// GC_ARMModel_47347719471423
// MI455X (gfx1250) — matrix operand order repaired and checked by reading, not compiled, not run on hardware
//
#include <hip/hip_runtime.h>
#include <stdint.h>

typedef __attribute__((ext_vector_type(16))) __bf16 v16bf;
typedef __attribute__((ext_vector_type(8)))  float  v8f;
typedef unsigned short u16;
typedef unsigned int   u32;

#define Bsz   4096
#define Fdim  64
#define Edim  16
#define Kdim  8
#define Odim  32
#define KO    256      // K*O
#define D0    4096     // K*O*E
#define HID   1024
#define EPSV  1e-5f

static __device__ __forceinline__ u16 f2bf(float x) {
    u32 u = __float_as_uint(x);
    u = (u + 0x7fffu + ((u >> 16) & 1u)) >> 16;
    return (u16)u;
}

// ---------------------------------------------------------------- embedding
// xe[b,f,e] = emb[x_id[b,f], e] * clip(x_value[b,f], 0.001, 1)
__global__ __launch_bounds__(256) void k_embed(const int* __restrict__ xid,
                                               const float* __restrict__ xval,
                                               const float* __restrict__ emb,
                                               float* __restrict__ xe) {
    int i = blockIdx.x * 256 + threadIdx.x;            // over B*F
    if (i >= Bsz * Fdim) return;
    float v = fminf(fmaxf(xval[i], 0.001f), 1.0f);
    const float4* ep = (const float4*)(emb + (size_t)xid[i] * Edim);
    float4* op = (float4*)(xe + (size_t)i * Edim);
#pragma unroll
    for (int j = 0; j < 4; ++j) {
        float4 q = ep[j];
        q.x *= v; q.y *= v; q.z *= v; q.w *= v;
        op[j] = q;
    }
}

// ------------------------------------------------- channel BN stats (train)
// X is (rows, C*G) row-major; channel ch covers G consecutive columns.
// scale[ch] = g*rstd ; shift[ch] = beta - mean*g*rstd. do_exp: stats of exp(X).
__global__ __launch_bounds__(256) void k_stats(const float* __restrict__ X,
                                               int rows, int C, int G,
                                               const float* __restrict__ gam,
                                               const float* __restrict__ bet,
                                               float* __restrict__ scale,
                                               float* __restrict__ shift,
                                               int do_exp) {
    int ch = blockIdx.x, t = threadIdx.x;
    float s = 0.f, ss = 0.f;
    for (int r = t; r < rows; r += 256) {
        const float* p = X + (size_t)r * C * G + (size_t)ch * G;
        for (int e = 0; e < G; ++e) {
            float v = p[e];
            if (do_exp) v = expf(v);
            s += v; ss += v * v;
        }
    }
    __shared__ float rs[256], rq[256];
    rs[t] = s; rq[t] = ss;
    __syncthreads();
    for (int o = 128; o > 0; o >>= 1) {
        if (t < o) { rs[t] += rs[t + o]; rq[t] += rq[t + o]; }
        __syncthreads();
    }
    if (t == 0) {
        float n = (float)rows * (float)G;
        float m = rs[0] / n;
        float var = rq[0] / n - m * m;
        float sc = gam[ch] * rsqrtf(var + EPSV);
        scale[ch] = sc;
        shift[ch] = bet[ch] - m * sc;
    }
}

// ------------------------------------------------------------- xsum (B,E)
__global__ __launch_bounds__(256) void k_xsum(const float* __restrict__ xe,
                                              float* __restrict__ xsum) {
    int i = blockIdx.x * 256 + threadIdx.x;            // B*E
    int b = i >> 4, e = i & 15;
    float s = 0.f;
    for (int f = 0; f < Fdim; ++f) s += xe[(size_t)b * (Fdim * Edim) + f * Edim + e];
    xsum[i] = s;
}

// --------------------------------------------- kq[k,o,x] = sum_y bil*Q  (tiny)
__global__ __launch_bounds__(256) void k_kq(const float* __restrict__ bil,
                                            const float* __restrict__ Q,
                                            float* __restrict__ kq) {
    int t = blockIdx.x * 256 + threadIdx.x;            // 4096 = k*512+o*16+x
    int k = t >> 9, o = (t >> 4) & 31, x = t & 15;
    const float* br = bil + ((size_t)k * 16 + x) * 16;
    const float* qr = Q + (size_t)(k * 32 + o) * 16;
    float s = 0.f;
#pragma unroll
    for (int y = 0; y < 16; ++y) s += br[y] * qr[y];
    kq[t] = s;
}

// --------- fused: scores + entmax15(50-iter bisection) + vals + arm GEMM ----
// one block per batch row b; thread t owns ko-row t. Avoids 256MB aw tensor.
__global__ __launch_bounds__(256, 1) void k_attn(const float* __restrict__ xe,
                                                 const float* __restrict__ xsum,
                                                 const float* __restrict__ esc,
                                                 const float* __restrict__ esh,
                                                 const float* __restrict__ kq,
                                                 const float* __restrict__ vals,
                                                 float* __restrict__ arm) {
    __shared__ float xe_s[Fdim * Edim];
    __shared__ float xen_s[Fdim * Edim];
    __shared__ float xs_s[Edim];
    int b = blockIdx.x, t = threadIdx.x;
    for (int i = t; i < Fdim * Edim; i += 256) {
        float v = xe[(size_t)b * (Fdim * Edim) + i];
        xe_s[i] = v;
        int f = i >> 4;
        xen_s[i] = expf(v) * esc[f] + esh[f];          // normalized exp(xe)
    }
    if (t < Edim) xs_s[t] = xsum[b * Edim + t];
    __syncthreads();

    float kqr[Edim];
#pragma unroll
    for (int e = 0; e < Edim; ++e) kqr[e] = kq[t * Edim + e];
    float gc = 0.f;
#pragma unroll
    for (int e = 0; e < Edim; ++e) gc += kqr[e] * xs_s[e];

    float s[Fdim];                                     // xs = 0.5*(sc+gc), in VGPRs
#pragma unroll
    for (int f = 0; f < Fdim; ++f) {
        float acc = gc;
#pragma unroll
        for (int e = 0; e < Edim; ++e) acc += kqr[e] * xe_s[f * Edim + e];
        s[f] = acc * 0.5f;
    }
    float mx = s[0];
#pragma unroll
    for (int f = 1; f < Fdim; ++f) mx = fmaxf(mx, s[f]);
    float tau_lo = mx - 1.0f;
    float dm = 0.875f;                                 // (mx-0.125) - (mx-1)
    float f_lo = -1.0f;
#pragma unroll
    for (int f = 0; f < Fdim; ++f) { float z = fmaxf(s[f] - tau_lo, 0.f); f_lo += z * z; }
    float tau_m = tau_lo;
#pragma unroll 1
    for (int it = 0; it < 50; ++it) {
        dm *= 0.5f;
        tau_m = tau_lo + dm;
        float fm = -1.0f;
#pragma unroll
        for (int f = 0; f < Fdim; ++f) { float z = fmaxf(s[f] - tau_m, 0.f); fm += z * z; }
        if (fm * f_lo >= 0.f) tau_lo = tau_m;
    }
    float psum = 0.f;
#pragma unroll
    for (int f = 0; f < Fdim; ++f) { float z = fmaxf(s[f] - tau_m, 0.f); s[f] = z * z; psum += s[f]; }
    float inv = 1.0f / psum;

    float armv[Edim];
#pragma unroll
    for (int e = 0; e < Edim; ++e) armv[e] = 0.f;
#pragma unroll
    for (int f = 0; f < Fdim; ++f) {
        float aw = s[f] * inv * vals[t * Fdim + f];
#pragma unroll
        for (int e = 0; e < Edim; ++e) armv[e] += aw * xen_s[f * Edim + e];
    }
#pragma unroll
    for (int e = 0; e < Edim; ++e)
        arm[(size_t)b * D0 + t * Edim + e] = armv[e];
}

// ------------------------------- normalize (+opt relu) -> bf16 / f32 -------
__global__ __launch_bounds__(256) void k_norm_bf16(const float* __restrict__ X,
                                                   const float* __restrict__ sc,
                                                   const float* __restrict__ sh,
                                                   u16* __restrict__ Y,
                                                   int n, int rowlen, int G, int do_relu) {
    int i = blockIdx.x * 256 + threadIdx.x;
    if (i >= n) return;
    int ch = (i % rowlen) / G;
    float v = X[i] * sc[ch] + sh[ch];
    if (do_relu) v = fmaxf(v, 0.f);
    Y[i] = f2bf(v);
}

__global__ __launch_bounds__(256) void k_norm_f32(const float* __restrict__ X,
                                                  const float* __restrict__ sc,
                                                  const float* __restrict__ sh,
                                                  float* __restrict__ Y,
                                                  int n, int rowlen, int G, int do_relu) {
    int i = blockIdx.x * 256 + threadIdx.x;
    if (i >= n) return;
    int ch = (i % rowlen) / G;
    float v = X[i] * sc[ch] + sh[ch];
    if (do_relu) v = fmaxf(v, 0.f);
    Y[i] = v;
}

__global__ __launch_bounds__(256) void k_cvt_bf16(const float* __restrict__ X,
                                                  u16* __restrict__ Y, int n) {
    int i = blockIdx.x * 256 + threadIdx.x;
    if (i >= n) return;
    Y[i] = f2bf(X[i]);
}

// ---------------- WMMA bf16 GEMM:  C[M,N] = A[M,K]·B[N,K]^T + bias ---------
// block = 256 threads = 8 waves (4 M x 2 N), block tile 128x128,
// wave tile 32x64: 2 A-frags x 4 B-frags -> 8 v_wmma_f32_16x16x32_bf16 per
// K-step from 12 b128 loads (~21 FLOP/B from registers), f32 accumulate.
// A fragment (16-bit 16x32, ISA table): lane m = l&15, K interleave
//   kb = kk + (l&16 ? 8:0); halves [kb..kb+7] and [kb+16..kb+23].
// B fragment (16-bit 32x16, row striped across lanes): lane n = l&15,
//   same K interleave as A: kb = kk + (l&16 ? 8:0); [kb..kb+7], [kb+16..kb+23].
__global__ __launch_bounds__(256) void k_gemm_wmma(const u16* __restrict__ A,
                                                   const u16* __restrict__ Bm,
                                                   const float* __restrict__ bias,
                                                   float* __restrict__ C,
                                                   int M, int N, int Kd) {
    const int lane = threadIdx.x & 31;
    const int wave = threadIdx.x >> 5;
    const int wn = wave & 1, wm = wave >> 1;
    const int m0 = blockIdx.y * 128 + wm * 32;
    const int n0 = blockIdx.x * 128 + wn * 64;
    const int lr  = lane & 15;
    const int kA  = (lane & 16) ? 8 : 0;               // A K-interleave select
    const int kB  = (lane & 16) ? 8 : 0;               // B K-interleave select

    v8f acc[2][4];
#pragma unroll
    for (int i = 0; i < 2; ++i)
#pragma unroll
        for (int j = 0; j < 4; ++j) acc[i][j] = (v8f){};

    union Frag { v16bf v; uint4 q[2]; };

    for (int kk = 0; kk < Kd; kk += 32) {
        Frag a[2], bfr[4];
#pragma unroll
        for (int i = 0; i < 2; ++i) {
            const u16* pa = A + (size_t)(m0 + 16 * i + lr) * Kd + kk + kA;
            a[i].q[0] = *(const uint4*)(pa);
            a[i].q[1] = *(const uint4*)(pa + 16);
        }
#pragma unroll
        for (int j = 0; j < 4; ++j) {
            const u16* pb = Bm + (size_t)(n0 + 16 * j + lr) * Kd + kk + kB;
            bfr[j].q[0] = *(const uint4*)(pb);
            bfr[j].q[1] = *(const uint4*)(pb + 16);
        }
#pragma unroll
        for (int i = 0; i < 2; ++i)
#pragma unroll
            for (int j = 0; j < 4; ++j)
                acc[i][j] = __builtin_amdgcn_wmma_f32_16x16x32_bf16(
                    false, a[i].v, false, bfr[j].v, (short)0, acc[i][j], false, false);
    }
    const int mb = (lane & 16) ? 8 : 0;
#pragma unroll
    for (int j = 0; j < 4; ++j) {
        const int nn = n0 + 16 * j + (lane & 15);
        const float bia = bias[nn];
#pragma unroll
        for (int i = 0; i < 2; ++i) {
            float* cp = C + (size_t)(m0 + 16 * i + mb) * N + nn;
#pragma unroll
            for (int r = 0; r < 8; ++r)
                cp[(size_t)r * N] = acc[i][j][r] + bia;
        }
    }
}

// -------------------------------------------- y[b] = h·wout + bout --------
__global__ __launch_bounds__(256) void k_out(const float* __restrict__ H,
                                             const float* __restrict__ wout,
                                             const float* __restrict__ bout,
                                             float* __restrict__ y) {
    int b = blockIdx.x, t = threadIdx.x;
    float acc = 0.f;
    for (int h = t; h < HID; h += 256) acc += H[(size_t)b * HID + h] * wout[h];
    __shared__ float r[256];
    r[t] = acc;
    __syncthreads();
    for (int o = 128; o > 0; o >>= 1) {
        if (t < o) r[t] += r[t + o];
        __syncthreads();
    }
    if (t == 0) y[b] = r[0] + bout[0];
}

// ===========================================================================
extern "C" void kernel_launch(void* const* d_in, const int* in_sizes, int n_in,
                              void* d_out, int out_size, void* d_ws, size_t ws_size,
                              hipStream_t stream) {
    const int*   x_id  = (const int*)d_in[0];
    const float* x_val = (const float*)d_in[1];
    const float* emb   = (const float*)d_in[2];
    const float* emb_g = (const float*)d_in[3];
    const float* emb_b = (const float*)d_in[4];
    const float* Q     = (const float*)d_in[5];
    const float* bil   = (const float*)d_in[6];
    const float* vals  = (const float*)d_in[7];
    const float* arm_g = (const float*)d_in[8];
    const float* arm_b = (const float*)d_in[9];
    const float* w1    = (const float*)d_in[10];
    const float* b1    = (const float*)d_in[11];
    const float* g1    = (const float*)d_in[12];
    const float* bt1   = (const float*)d_in[13];
    const float* w2    = (const float*)d_in[14];
    const float* b2    = (const float*)d_in[15];
    const float* g2    = (const float*)d_in[16];
    const float* bt2   = (const float*)d_in[17];
    const float* wout  = (const float*)d_in[18];
    const float* bout  = (const float*)d_in[19];

    char* ws = (char*)d_ws;
    size_t off = 0;
    auto carve = [&](size_t bytes) { char* p = ws + off; off = (off + bytes + 255) & ~(size_t)255; return p; };

    float* xe       = (float*)carve((size_t)Bsz * Fdim * Edim * 4);   // 16 MB
    float* xsum     = (float*)carve((size_t)Bsz * Edim * 4);
    float* esc      = (float*)carve(Fdim * 4);
    float* esh      = (float*)carve(Fdim * 4);
    float* kq       = (float*)carve(D0 * 4);
    float* arm_raw  = (float*)carve((size_t)Bsz * D0 * 4);            // 64 MB
    float* armsc    = (float*)carve(KO * 4);
    float* armsh    = (float*)carve(KO * 4);
    u16*   armn     = (u16*)carve((size_t)Bsz * D0 * 2);              // 32 MB
    u16*   w1b      = (u16*)carve((size_t)HID * D0 * 2);              // 8 MB
    u16*   w2b      = (u16*)carve((size_t)HID * HID * 2);             // 2 MB
    float* h1sc     = (float*)carve(HID * 4);
    float* h1sh     = (float*)carve(HID * 4);
    u16*   h1n      = (u16*)carve((size_t)Bsz * HID * 2);             // 8 MB
    float* h2sc     = (float*)carve(HID * 4);
    float* h2sh     = (float*)carve(HID * 4);
    // alias h1/h2/h2n into arm_raw's 64MB region (dead after armn is built)
    float* h1  = arm_raw;                                             // 16 MB
    float* h2  = arm_raw + (size_t)Bsz * HID;                         // 16 MB
    float* h2n = arm_raw + (size_t)2 * Bsz * HID;                     // 16 MB

    // 1) embedding gather * clipped value
    k_embed<<<(Bsz * Fdim) / 256, 256, 0, stream>>>(x_id, x_val, emb, xe);
    // 2) BN stats of exp(xe) per F channel (exp recomputed on the fly)
    k_stats<<<Fdim, 256, 0, stream>>>(xe, Bsz, Fdim, Edim, emb_g, emb_b, esc, esh, 1);
    // 3) xsum[b,e]
    k_xsum<<<(Bsz * Edim) / 256, 256, 0, stream>>>(xe, xsum);
    // 4) kq = einsum(bil, Q)
    k_kq<<<D0 / 256, 256, 0, stream>>>(bil, Q, kq);
    // 5) fused scores + entmax15 + vals + arm contraction
    k_attn<<<Bsz, 256, 0, stream>>>(xe, xsum, esc, esh, kq, vals, arm_raw);
    // 6) arm BN stats + 7) normalize -> bf16
    k_stats<<<KO, 256, 0, stream>>>(arm_raw, Bsz, KO, Edim, arm_g, arm_b, armsc, armsh, 0);
    k_norm_bf16<<<(Bsz * D0) / 256, 256, 0, stream>>>(arm_raw, armsc, armsh, armn,
                                                      Bsz * D0, D0, Edim, 0);
    // 8) weights -> bf16
    k_cvt_bf16<<<(HID * D0) / 256, 256, 0, stream>>>(w1, w1b, HID * D0);
    k_cvt_bf16<<<(HID * HID) / 256, 256, 0, stream>>>(w2, w2b, HID * HID);
    // 9) h1 = armn · w1^T + b1   (WMMA, 128x128 block tile)
    k_gemm_wmma<<<dim3(HID / 128, Bsz / 128), 256, 0, stream>>>(armn, w1b, b1, h1, Bsz, HID, D0);
    // 10-11) BN + relu -> bf16
    k_stats<<<HID, 256, 0, stream>>>(h1, Bsz, HID, 1, g1, bt1, h1sc, h1sh, 0);
    k_norm_bf16<<<(Bsz * HID) / 256, 256, 0, stream>>>(h1, h1sc, h1sh, h1n, Bsz * HID, HID, 1, 1);
    // 12) h2 = h1n · w2^T + b2   (WMMA)
    k_gemm_wmma<<<dim3(HID / 128, Bsz / 128), 256, 0, stream>>>(h1n, w2b, b2, h2, Bsz, HID, HID);
    // 13-14) BN + relu -> f32
    k_stats<<<HID, 256, 0, stream>>>(h2, Bsz, HID, 1, g2, bt2, h2sc, h2sh, 0);
    k_norm_f32<<<(Bsz * HID) / 256, 256, 0, stream>>>(h2, h2sc, h2sh, h2n, Bsz * HID, HID, 1, 1);
    // 15) output row
    k_out<<<Bsz, 256, 0, stream>>>(h2n, wout, bout, (float*)d_out);
}
